// DGCNN_53197464929043
// MI455X (gfx1250) — matrix operand order repaired and checked by reading, not compiled, not run on hardware
//
#include <hip/hip_runtime.h>
#include <hip/hip_bf16.h>

// ---------------- WMMA types ----------------
typedef __attribute__((ext_vector_type(16))) __bf16 v16bf;
typedef __attribute__((ext_vector_type(8)))  float  v8f;

union FragB16 { v16bf v; uint4 q[2]; unsigned int u[8]; };
union FragAcc { v8f v; float f[8]; };

__device__ __forceinline__ unsigned short f2bf(float f) {
    unsigned int u = __float_as_uint(f);
    unsigned int lsb = (u >> 16) & 1u;
    u += 0x7FFFu + lsb;               // round-to-nearest-even
    return (unsigned short)(u >> 16);
}

#define NPTS 4096
#define BATCH 8
#define NODES (BATCH * NPTS)
#define KNN 5
#define RSQRT5 0.44721359549995793f

// ---------------- zero init ----------------
__global__ void zero_u32_kernel(unsigned int* p, int n) {
    int i = blockIdx.x * blockDim.x + threadIdx.x;
    if (i < n) p[i] = 0u;
}

// ---------------- centroid + distance-to-centroid ----------------
__global__ __launch_bounds__(256) void centroid_kernel(const float* __restrict__ pts,
                                                       float* __restrict__ dcen) {
    int b = blockIdx.x, tid = threadIdx.x;
    __shared__ float sx[256], sy[256], sz[256];
    const float* px = pts + (size_t)(b * 3 + 0) * NPTS;
    const float* py = pts + (size_t)(b * 3 + 1) * NPTS;
    const float* pz = pts + (size_t)(b * 3 + 2) * NPTS;
    float ax = 0.f, ay = 0.f, az = 0.f;
    for (int i = tid; i < NPTS; i += 256) { ax += px[i]; ay += py[i]; az += pz[i]; }
    sx[tid] = ax; sy[tid] = ay; sz[tid] = az;
    __syncthreads();
    for (int s = 128; s > 0; s >>= 1) {
        if (tid < s) { sx[tid] += sx[tid + s]; sy[tid] += sy[tid + s]; sz[tid] += sz[tid + s]; }
        __syncthreads();
    }
    float cx = sx[0] * (1.f / NPTS), cy = sy[0] * (1.f / NPTS), cz = sz[0] * (1.f / NPTS);
    for (int i = tid; i < NPTS; i += 256) {
        float dx = px[i] - cx, dy = py[i] - cy, dz = pz[i] - cz;
        dcen[b * NPTS + i] = dx * dx + dy * dy + dz * dz;
    }
}

// ---------------- rank (stable argsort position) + scatter coords to bf16 [32768,32] ----------------
__global__ __launch_bounds__(256) void rank_scatter_kernel(const float* __restrict__ pts,
                                                           const float* __restrict__ dcen,
                                                           unsigned short* __restrict__ X32) {
    int b = blockIdx.x >> 4;
    int i = ((blockIdx.x & 15) << 8) + threadIdx.x;
    float di = dcen[b * NPTS + i];
    __shared__ float sd[256];
    int rank = 0;
    for (int tile = 0; tile < 16; ++tile) {
        sd[threadIdx.x] = dcen[b * NPTS + tile * 256 + threadIdx.x];
        __syncthreads();
        int jbase = tile * 256;
        for (int jj = 0; jj < 256; ++jj) {
            float dj = sd[jj];
            int j = jbase + jj;
            rank += (dj < di) || (dj == di && j < i);
        }
        __syncthreads();
    }
    size_t row = (size_t)b * NPTS + rank;
    #pragma unroll
    for (int c = 0; c < 3; ++c)
        X32[row * 32 + c] = f2bf(pts[(size_t)(b * 3 + c) * NPTS + i]);
}

// ---------------- KNN (k=5, self included) + deg_out counting ----------------
__global__ __launch_bounds__(256) void knn_kernel(const float* __restrict__ pts,
                                                  int* __restrict__ nbr,
                                                  int* __restrict__ degout) {
    int b = blockIdx.x >> 4;
    int i = ((blockIdx.x & 15) << 8) + threadIdx.x;
    const float* px = pts + (size_t)(b * 3 + 0) * NPTS;
    const float* py = pts + (size_t)(b * 3 + 1) * NPTS;
    const float* pz = pts + (size_t)(b * 3 + 2) * NPTS;
    float xi = px[i], yi = py[i], zi = pz[i];
    __shared__ float sx[512], sy[512], sz[512];
    float bd[KNN]; int bi[KNN];
    #pragma unroll
    for (int t = 0; t < KNN; ++t) { bd[t] = 3.4e38f; bi[t] = 0; }
    for (int tile = 0; tile < NPTS / 512; ++tile) {
        #pragma unroll
        for (int t = 0; t < 2; ++t) {
            int idx = tile * 512 + t * 256 + threadIdx.x;
            sx[t * 256 + threadIdx.x] = px[idx];
            sy[t * 256 + threadIdx.x] = py[idx];
            sz[t * 256 + threadIdx.x] = pz[idx];
        }
        __syncthreads();
        for (int jj = 0; jj < 512; ++jj) {
            float dx = sx[jj] - xi, dy = sy[jj] - yi, dz = sz[jj] - zi;
            float d = fmaf(dx, dx, fmaf(dy, dy, dz * dz));
            if (d < bd[KNN - 1]) {
                int j = tile * 512 + jj;
                bd[KNN - 1] = d; bi[KNN - 1] = j;
                #pragma unroll
                for (int t = KNN - 1; t > 0; --t) {
                    if (bd[t] < bd[t - 1]) {
                        float td = bd[t]; bd[t] = bd[t - 1]; bd[t - 1] = td;
                        int ti = bi[t]; bi[t] = bi[t - 1]; bi[t - 1] = ti;
                    }
                }
            }
        }
        __syncthreads();
    }
    int node = b * NPTS + i;
    #pragma unroll
    for (int t = 0; t < KNN; ++t) {
        int g = b * NPTS + bi[t];
        nbr[node * KNN + t] = g;
        atomicAdd(&degout[g], 1);
    }
}

__global__ void normout_kernel(const int* __restrict__ degout, float* __restrict__ normout) {
    int i = blockIdx.x * blockDim.x + threadIdx.x;
    if (i < NODES) normout[i] = rsqrtf(fmaxf((float)degout[i], 1.0f));
}

// ---------------- weight prepack: fp32 -> bf16 in WMMA B-fragment layout ----------------
// B-matrix 32x16 (16-bit) per ISA: lane h = lane>>4, n = lane&15 ; VGPR v holds K = kb, kb+1 with kb = v<4 ? 2v+8h : 16+2(v-4)+8h
// Storage: dst[tile*256 + lane*8 + v] so each lane's 8 dwords are contiguous (2x b128 loads).
__global__ __launch_bounds__(256) void pack_w_kernel(const float* __restrict__ W,
                                                     unsigned int* __restrict__ dst,
                                                     int K, int N, int srcK, int transposed) {
    int gid = blockIdx.x * blockDim.x + threadIdx.x;
    int lane = gid & 31;
    int tile = gid >> 5;
    int nTiles16 = N >> 4;
    int totTiles = (K >> 5) * nTiles16;
    if (tile >= totTiles) return;
    int kt = tile / nTiles16, nt = tile - kt * nTiles16;
    int h = lane >> 4, nn = lane & 15;
    int n = nt * 16 + nn;
    #pragma unroll
    for (int v = 0; v < 8; ++v) {
        int k0 = kt * 32 + (v < 4 ? 2 * v + 8 * h : 16 + 2 * (v - 4) + 8 * h);
        float f0 = 0.f, f1 = 0.f;
        if (k0 < srcK)     f0 = transposed ? W[(size_t)n * srcK + k0]     : W[(size_t)k0 * N + n];
        if (k0 + 1 < srcK) f1 = transposed ? W[(size_t)n * srcK + k0 + 1] : W[(size_t)(k0 + 1) * N + n];
        dst[(size_t)tile * 256 + lane * 8 + v] =
            (unsigned int)f2bf(f0) | ((unsigned int)f2bf(f1) << 16);
    }
}

// ---------------- generic WMMA bf16 GEMM, 16x64 per wave ----------------
// Software-pipelined: fragments for K-step kt+1 are loaded BEFORE the 4 WMMAs
// of step kt execute, so the loadcnt wait before the WMMA group drains only to
// "next-step loads outstanding" instead of zero.
template<bool HASROW, bool HASCOL, bool RELU, bool OUTF, bool OUTBF, bool POOL>
__global__ __launch_bounds__(32) void gemm_wmma_kernel(
        const unsigned short* __restrict__ A, const unsigned int* __restrict__ Bp,
        int M, int N, int K,
        const float* __restrict__ rowScale,
        const float* __restrict__ colScale, const float* __restrict__ colBias,
        float* __restrict__ outF, unsigned short* __restrict__ outBf,
        int* __restrict__ poolMax) {
    const int lane = threadIdx.x;
    const int rt = blockIdx.x;      // 16-row stripe
    const int cg = blockIdx.y;      // 64-col group
    const int h  = lane >> 4;
    const int mn = lane & 15;
    const int nTiles16 = N >> 4;

    v8f acc[4];
    #pragma unroll
    for (int t = 0; t < 4; ++t)
        acc[t] = (v8f){0.f, 0.f, 0.f, 0.f, 0.f, 0.f, 0.f, 0.f};

    const unsigned short* arow = A + (size_t)(rt * 16 + mn) * K;
    const int kTiles = K >> 5;

    // A-matrix 16x32 (16-bit) per ISA: v<4 -> K=2v+8h ; v>=4 -> K=16+2(v-4)+8h
    // -> each half is 16 contiguous bytes: two b128 loads.
    auto loadFrags = [&](int kt, FragB16& a, FragB16 (&b)[4]) {
        const int kbase = kt * 32 + 8 * h;
        a.q[0] = *(const uint4*)(arow + kbase);
        a.q[1] = *(const uint4*)(arow + kbase + 16);
        const unsigned int* bt = Bp + ((size_t)kt * nTiles16 + cg * 4) * 256 + lane * 8;
        #pragma unroll
        for (int t = 0; t < 4; ++t) {
            b[t].q[0] = *(const uint4*)(bt + t * 256);
            b[t].q[1] = *(const uint4*)(bt + t * 256 + 4);
        }
    };

    FragB16 aCur, bCur[4];
    loadFrags(0, aCur, bCur);
    for (int kt = 0; kt + 1 < kTiles; ++kt) {
        FragB16 aNxt, bNxt[4];
        loadFrags(kt + 1, aNxt, bNxt);           // prefetch next K-step
        #pragma unroll
        for (int t = 0; t < 4; ++t) {            // consume current (already resident)
            acc[t] = __builtin_amdgcn_wmma_f32_16x16x32_bf16(
                false, aCur.v, false, bCur[t].v, (short)0, acc[t], false, false);
        }
        aCur = aNxt;
        #pragma unroll
        for (int t = 0; t < 4; ++t) bCur[t] = bNxt[t];
    }
    #pragma unroll
    for (int t = 0; t < 4; ++t) {                // epilogue K-step
        acc[t] = __builtin_amdgcn_wmma_f32_16x16x32_bf16(
            false, aCur.v, false, bCur[t].v, (short)0, acc[t], false, false);
    }

    float rs[8];
    #pragma unroll
    for (int v = 0; v < 8; ++v)
        rs[v] = HASROW ? rowScale[rt * 16 + v + 8 * h] : 1.0f;

    #pragma unroll
    for (int t = 0; t < 4; ++t) {
        FragAcc ac; ac.v = acc[t];
        int col = cg * 64 + t * 16 + mn;
        float csc = HASCOL ? colScale[col] : 1.0f;
        float cb  = HASCOL ? colBias[col]  : 0.0f;
        float pmax = 0.0f;
        #pragma unroll
        for (int v = 0; v < 8; ++v) {
            int r = rt * 16 + v + 8 * h;
            float val = ac.f[v] * rs[v] * csc + cb;
            if (RELU) val = fmaxf(val, 0.0f);
            size_t off = (size_t)r * N + col;
            if (OUTF)  outF[off]  = val;
            if (OUTBF) outBf[off] = f2bf(val);
            if (POOL)  pmax = fmaxf(pmax, val);
        }
        if (POOL) atomicMax(&poolMax[col], __float_as_int(pmax));  // vals >= 0
    }
}

// ---------------- GCN neighbor gather-sum + norm_in + bias ----------------
__global__ __launch_bounds__(256) void gcn_gather_kernel(const float* __restrict__ hw,
                                                         const int* __restrict__ nbr,
                                                         const float* __restrict__ bias,
                                                         unsigned short* __restrict__ outBf,
                                                         int C, int relu) {
    int gid = blockIdx.x * 256 + threadIdx.x;
    int node = gid / C;
    int c = gid - node * C;
    const int* nb = nbr + (size_t)node * KNN;
    float s = 0.f;
    #pragma unroll
    for (int t = 0; t < KNN; ++t) s += hw[(size_t)nb[t] * C + c];
    float v = s * RSQRT5 + bias[c];
    if (relu) v = fmaxf(v, 0.0f);
    outBf[gid] = f2bf(v);
}

// ---------------- final tiny GEMV: out[128] = pooled[256] @ fc3_w.T + fc3_b ----------------
__global__ __launch_bounds__(128) void final_kernel(const int* __restrict__ pool,
                                                    const float* __restrict__ w,
                                                    const float* __restrict__ b,
                                                    float* __restrict__ out) {
    int o = threadIdx.x;
    float s = 0.f;
    for (int n = 0; n < 256; ++n)
        s += __int_as_float(pool[n]) * w[o * 256 + n];
    out[o] = s + b[o];
}

// ================= host launcher =================
extern "C" void kernel_launch(void* const* d_in, const int* in_sizes, int n_in,
                              void* d_out, int out_size, void* d_ws, size_t ws_size,
                              hipStream_t stream) {
    (void)in_sizes; (void)n_in; (void)out_size; (void)ws_size;
    const float* pts     = (const float*)d_in[0];
    const float* conv_w1 = (const float*)d_in[1];
    const float* bn1_g   = (const float*)d_in[2];
    const float* bn1_b   = (const float*)d_in[3];
    const float* conv_w2 = (const float*)d_in[4];
    const float* bn2_g   = (const float*)d_in[5];
    const float* bn2_b   = (const float*)d_in[6];
    const float* conv_w3 = (const float*)d_in[7];
    const float* bn3_g   = (const float*)d_in[8];
    const float* bn3_b   = (const float*)d_in[9];
    const float* conv_w4 = (const float*)d_in[10];
    const float* bn4_g   = (const float*)d_in[11];
    const float* bn4_b   = (const float*)d_in[12];
    const float* gc1_w   = (const float*)d_in[13];
    const float* gc1_b   = (const float*)d_in[14];
    const float* gc2_w   = (const float*)d_in[15];
    const float* gc2_b   = (const float*)d_in[16];
    const float* gc3_w   = (const float*)d_in[17];
    const float* gc3_b   = (const float*)d_in[18];
    const float* fc1_w   = (const float*)d_in[19];
    const float* bnf1_g  = (const float*)d_in[20];
    const float* bnf1_b  = (const float*)d_in[21];
    const float* fc2_w   = (const float*)d_in[22];
    const float* bnf2_g  = (const float*)d_in[23];
    const float* bnf2_b  = (const float*)d_in[24];
    const float* fc3_w   = (const float*)d_in[25];
    const float* fc3_b   = (const float*)d_in[26];
    float* out = (float*)d_out;

    // workspace carve-out (256B aligned)
    char* base = (char*)d_ws;
    size_t off = 0;
    auto alloc = [&](size_t bytes) -> void* {
        void* p = base + off;
        off += (bytes + 255) & ~(size_t)255;
        return p;
    };
    unsigned short* X32    = (unsigned short*)alloc((size_t)NODES * 32 * 2);
    float*          dcen   = (float*)alloc((size_t)NODES * 4);
    int*            nbr    = (int*)alloc((size_t)NODES * KNN * 4);
    int*            degout = (int*)alloc((size_t)NODES * 4);
    float*          nrmout = (float*)alloc((size_t)NODES * 4);
    int*            pool   = (int*)alloc(256 * 4);
    unsigned int* w1p = (unsigned int*)alloc((size_t)32  * 64  * 2);
    unsigned int* w2p = (unsigned int*)alloc((size_t)64  * 64  * 2);
    unsigned int* w3p = (unsigned int*)alloc((size_t)64  * 64  * 2);
    unsigned int* w4p = (unsigned int*)alloc((size_t)64  * 128 * 2);
    unsigned int* g1p = (unsigned int*)alloc((size_t)128 * 128 * 2);
    unsigned int* g2p = (unsigned int*)alloc((size_t)128 * 128 * 2);
    unsigned int* g3p = (unsigned int*)alloc((size_t)128 * 256 * 2);
    unsigned int* f1p = (unsigned int*)alloc((size_t)256 * 512 * 2);
    unsigned int* f2p = (unsigned int*)alloc((size_t)512 * 256 * 2);
    unsigned short* actA = (unsigned short*)alloc((size_t)NODES * 512 * 2);
    unsigned short* actB = (unsigned short*)alloc((size_t)NODES * 512 * 2);
    float*          hw   = (float*)alloc((size_t)NODES * 256 * 4);

    // ---- zero init (every call: atomics + padded cols must start at 0) ----
    {
        int nX = NODES * 32 / 2;  // dwords in X32
        zero_u32_kernel<<<(nX + 255) / 256, 256, 0, stream>>>((unsigned int*)X32, nX);
        zero_u32_kernel<<<(NODES + 255) / 256, 256, 0, stream>>>((unsigned int*)degout, NODES);
        zero_u32_kernel<<<1, 256, 0, stream>>>((unsigned int*)pool, 256);
    }

    // ---- weight prepack ----
    auto packLaunch = [&](const float* W, unsigned int* dst, int K, int N, int srcK, int tr) {
        int threads = (K >> 5) * (N >> 4) * 32;
        pack_w_kernel<<<(threads + 255) / 256, 256, 0, stream>>>(W, dst, K, N, srcK, tr);
    };
    packLaunch(conv_w1, w1p, 32,  64,  3,   1);   // [64,3]   used as h@w.T
    packLaunch(conv_w2, w2p, 64,  64,  64,  1);
    packLaunch(conv_w3, w3p, 64,  64,  64,  1);
    packLaunch(conv_w4, w4p, 64,  128, 64,  1);
    packLaunch(gc1_w,   g1p, 128, 128, 128, 0);   // [in,out] used as h@w
    packLaunch(gc2_w,   g2p, 128, 128, 128, 0);
    packLaunch(gc3_w,   g3p, 128, 256, 128, 0);
    packLaunch(fc1_w,   f1p, 256, 512, 256, 1);
    packLaunch(fc2_w,   f2p, 512, 256, 512, 1);

    // ---- sort-by-centroid + scatter bf16 coords; KNN graph ----
    centroid_kernel<<<BATCH, 256, 0, stream>>>(pts, dcen);
    rank_scatter_kernel<<<BATCH * (NPTS / 256), 256, 0, stream>>>(pts, dcen, X32);
    knn_kernel<<<BATCH * (NPTS / 256), 256, 0, stream>>>(pts, nbr, degout);
    normout_kernel<<<(NODES + 255) / 256, 256, 0, stream>>>(degout, nrmout);

    // Epilogue variants (compile-time specialized):
    //  MLP/head: colScale+bias, ReLU, bf16 out
    //  GCN mm  : rowScale, f32 out
    //  FC2     : colScale+bias, ReLU, fused column-max pool
    auto gemmMLP = [&](const unsigned short* A, const unsigned int* Bp, int N, int K,
                       const float* csc, const float* cbi, unsigned short* oB) {
        dim3 grid(NODES / 16, N / 64);
        gemm_wmma_kernel<false, true, true, false, true, false>
            <<<grid, 32, 0, stream>>>(A, Bp, NODES, N, K, nullptr, csc, cbi,
                                      nullptr, oB, nullptr);
    };
    auto gemmGCN = [&](const unsigned short* A, const unsigned int* Bp, int N, int K,
                       const float* rsc, float* oF) {
        dim3 grid(NODES / 16, N / 64);
        gemm_wmma_kernel<true, false, false, true, false, false>
            <<<grid, 32, 0, stream>>>(A, Bp, NODES, N, K, rsc, nullptr, nullptr,
                                      oF, nullptr, nullptr);
    };
    auto gemmPOOL = [&](const unsigned short* A, const unsigned int* Bp, int N, int K,
                        const float* csc, const float* cbi, int* pm) {
        dim3 grid(NODES / 16, N / 64);
        gemm_wmma_kernel<false, true, true, false, false, true>
            <<<grid, 32, 0, stream>>>(A, Bp, NODES, N, K, nullptr, csc, cbi,
                                      nullptr, nullptr, pm);
    };

    // ---- per-point MLP: 3(pad32)->64->64->64->128, BN+ReLU ----
    gemmMLP(X32,  w1p, 64,  32, bn1_g, bn1_b, actA);
    gemmMLP(actA, w2p, 64,  64, bn2_g, bn2_b, actB);
    gemmMLP(actB, w3p, 64,  64, bn3_g, bn3_b, actA);
    gemmMLP(actA, w4p, 128, 64, bn4_g, bn4_b, actB);   // feat

    // ---- GCN layers: (h@W)*norm_out ; gather-sum ; *rsqrt(5)+bias (+relu) ----
    gemmGCN(actB, g1p, 128, 128, nrmout, hw);
    gcn_gather_kernel<<<NODES * 128 / 256, 256, 0, stream>>>(hw, nbr, gc1_b, actA, 128, 1);
    gemmGCN(actA, g2p, 128, 128, nrmout, hw);
    gcn_gather_kernel<<<NODES * 128 / 256, 256, 0, stream>>>(hw, nbr, gc2_b, actB, 128, 1);
    gemmGCN(actB, g3p, 256, 128, nrmout, hw);
    gcn_gather_kernel<<<NODES * 256 / 256, 256, 0, stream>>>(hw, nbr, gc3_b, actA, 256, 0);

    // ---- head: fc1 (BN+ReLU), fc2 (BN+ReLU fused with global column-max pooling) ----
    gemmMLP(actA, f1p, 512, 256, bnf1_g, bnf1_b, actB);
    gemmPOOL(actB, f2p, 256, 512, bnf2_g, bnf2_b, pool);

    // ---- out[128] = pooled @ fc3_w.T + fc3_b ----
    final_kernel<<<1, 128, 0, stream>>>(pool, fc3_w, fc3_b, out);
}
